// SpaceCrossAttention_62706522521715
// MI455X (gfx1250) — matrix operand order repaired and checked by reading, not compiled, not run on hardware
//
#include <hip/hip_runtime.h>
#include <hip/hip_bf16.h>

#define Bsz  4
#define Cdim 256
#define Ndim 4096
#define Hdim 128

typedef __attribute__((ext_vector_type(16))) __bf16 v16bf;
typedef __attribute__((ext_vector_type(8)))  __bf16 v8bf;
typedef __attribute__((ext_vector_type(8)))  float  v8f;
typedef unsigned int v4u __attribute__((ext_vector_type(4)));
typedef int          v8i __attribute__((ext_vector_type(8)));
typedef int          v4i __attribute__((ext_vector_type(4)));

#if defined(__has_builtin)
#if __has_builtin(__builtin_amdgcn_tensor_load_to_lds) && __has_builtin(__builtin_amdgcn_s_wait_tensorcnt)
#define HAVE_TDM 1
#endif
#endif

// LDS tile layout: 16 rows x 256B, padded to 272B/row (TDM pad: 16B per 256B)
// -> lane l15's ds reads land on banks 4*l15 .. 4*l15+3 : conflict-free.
#define TILE_LDS_BYTES 4352   // 16 * 272

// ---------------------------------------------------------------------------
// Kernel 1: projection  mapped[b,n,h] = sum_c x[b,c,n] * W[h,c] + bias[h]
// stored as bf16, row-major [B*N, H].
// ---------------------------------------------------------------------------
__global__ __launch_bounds__(256) void map_project_kernel(
    const float* __restrict__ x, const float* __restrict__ W,
    const float* __restrict__ bias, __hip_bfloat16* __restrict__ out)
{
    __shared__ float xs[Cdim][32];
    const int b  = blockIdx.y;
    const int n0 = blockIdx.x * 32;
    const int t  = threadIdx.x;
    const float* xb = x + (size_t)b * Cdim * Ndim;

    const int csub = t >> 5, noff = t & 31;
    #pragma unroll
    for (int cb = 0; cb < Cdim; cb += 8)
        xs[cb + csub][noff] = xb[(size_t)(cb + csub) * Ndim + n0 + noff];
    __syncthreads();

    const int h  = t & 127;
    const int ng = t >> 7;
    float acc[16];
    const float bh = bias[h];
    #pragma unroll
    for (int i = 0; i < 16; ++i) acc[i] = bh;

    const float4* wr = (const float4*)(W + (size_t)h * Cdim);
    for (int c4 = 0; c4 < Cdim / 4; ++c4) {
        const float4 w = wr[c4];
        const int c = c4 * 4;
        #pragma unroll
        for (int i = 0; i < 16; ++i) {
            const int nf = ng * 16 + i;
            acc[i] += xs[c+0][nf]*w.x + xs[c+1][nf]*w.y
                    + xs[c+2][nf]*w.z + xs[c+3][nf]*w.w;
        }
    }
    #pragma unroll
    for (int i = 0; i < 16; ++i)
        out[(size_t)((b * Ndim) + n0 + ng * 16 + i) * Hdim + h] =
            __float2bfloat16(acc[i]);
}

#ifdef HAVE_TDM
// 1-D TDM descriptor: copy one contiguous 4KB B tile (512 x 8B elements) from
// global to LDS, inserting 16B of padding after every 256B (row) stored.
__device__ __forceinline__ void tdm_load_tile(unsigned lds_off, unsigned long long gaddr)
{
    v4u g0;
    g0.x = 1u;                                         // count=1, user mode
    g0.y = lds_off;                                    // lds_addr
    g0.z = (unsigned)gaddr;                            // global_addr[31:0]
    g0.w = (unsigned)((gaddr >> 32) & 0x01FFFFFFu) | (2u << 30);  // ga[56:32] | type=2
    v8i g1;
    g1[0] = (3 << 16)            // data_size = 8B
          | (1 << 20)            // pad_enable
          | (5 << 22)            // pad_interval: 64 dwords (256B)
          | (3 << 25);           // pad_amount: 4 dwords (16B)
    g1[1] = (int)(512u << 16);   // tensor_dim0 = 512 (bits 79:48)
    g1[2] = (1 << 16);           // tensor_dim1 = 1  (bits 111:80)
    g1[3] = (int)(512u << 16);   // tile_dim0 = 512  (bits 127:112)
    g1[4] = 0;                   // tile_dim1 = 0 (1-D), tile_dim2 = 0
    g1[5] = 512;                 // tensor_dim0_stride = 512
    g1[6] = 0;
    g1[7] = 0;
    const v4i z4 = {0, 0, 0, 0};
#if __clang_major__ >= 23
    const v8i z8 = {0, 0, 0, 0, 0, 0, 0, 0};
    __builtin_amdgcn_tensor_load_to_lds(g0, g1, z4, z4, z8, 0);
#else
    __builtin_amdgcn_tensor_load_to_lds(g0, g1, z4, z4, 0);
#endif
}
#endif

// ---------------------------------------------------------------------------
// Kernel 2: scores + softmax diagonal, flash-style.
// Block = 8 waves = 128 rows of one batch. The B panel is staged through LDS
// tile-by-tile (groups of 4 tiles, double buffered) so all 8 waves share each
// 4KB tile: 8x less L2 traffic. Staging uses the Tensor Data Mover
// (tensor_load_to_lds + s_wait_tensorcnt) when available.
// Per tile: 8 ds_loads, 4x v_wmma_f32_16x16x32_bf16, lane-local online softmax
// (single v_exp per element, no cross-lane ops). States merge once at the end.
// ---------------------------------------------------------------------------
__global__ __launch_bounds__(256) void score_diag_kernel(
    const __hip_bfloat16* __restrict__ ma_, const __hip_bfloat16* __restrict__ mb_,
    float* __restrict__ diag)
{
    __shared__ __align__(16) char sbuf[2][4][TILE_LDS_BYTES];   // 34 KB

    const int lane = threadIdx.x & 31;
    const int wave = threadIdx.x >> 5;
    const int b       = blockIdx.x >> 5;            // 32 blocks per batch
    const int rowbase = (blockIdx.x & 31) * 128;
    const int n0      = rowbase + wave * 16;        // this wave's 16 rows

    const __bf16* A  = (const __bf16*)ma_ + (size_t)b * Ndim * Hdim;
    const __bf16* Bm = (const __bf16*)mb_ + (size_t)b * Ndim * Hdim;
    const int half = lane >> 4, l15 = lane & 15;

    // Resident A operand (contiguous k order: k = 16*half + s).
    v16bf areg[4];
    {
        const __bf16* Ar = A + (size_t)(n0 + l15) * Hdim + 16 * half;
        #pragma unroll
        for (int kc = 0; kc < 4; ++kc) {
            v8bf lo = *(const v8bf*)(Ar + kc * 32);
            v8bf hi = *(const v8bf*)(Ar + kc * 32 + 8);
            areg[kc] = __builtin_shufflevector(lo, hi,
                        0,1,2,3,4,5,6,7,8,9,10,11,12,13,14,15);
        }
    }

    float m_l[8], s_l[8];
    bool  dlane[8];
    #pragma unroll
    for (int j = 0; j < 8; ++j) {
        m_l[j] = -INFINITY; s_l[j] = 0.f;
        dlane[j] = (l15 == j + 8 * half);
    }

    // Consume one 16x128 B tile that sits in LDS (padded rows, 272B apart).
    auto consume_tile = [&](const char* tbase) {
        v8f acc = {0.f,0.f,0.f,0.f,0.f,0.f,0.f,0.f};
        const char* rp = tbase + l15 * 272 + half * 32;
        #pragma unroll
        for (int kc = 0; kc < 4; ++kc) {
            v8bf b0 = *(const v8bf*)(rp + kc * 64);
            v8bf b1 = *(const v8bf*)(rp + kc * 64 + 16);
            v16bf bv = __builtin_shufflevector(b0, b1,
                         0,1,2,3,4,5,6,7,8,9,10,11,12,13,14,15);
            acc = __builtin_amdgcn_wmma_f32_16x16x32_bf16(
                      false, areg[kc], false, bv, (short)0, acc, false, false);
        }
        #pragma unroll
        for (int j = 0; j < 8; ++j) {       // lane-local online softmax, 1 exp
            const float s  = acc[j];
            const float mo = m_l[j];
            const float t  = __expf(-fabsf(s - mo));
            const bool  gt = (s > mo);
            s_l[j] = fmaf(s_l[j], gt ? t : 1.f, gt ? 1.f : t);
            m_l[j] = fmaxf(mo, s);
        }
    };

    const int NGROUPS = (Ndim / 16) / 4;            // 64 groups of 4 tiles

#ifdef HAVE_TDM
    auto issue_group = [&](int g, int sel) {
        #pragma unroll
        for (int i = 0; i < 4; ++i) {
            const int t = g * 4 + i;
            tdm_load_tile((unsigned)(size_t)&sbuf[sel][i][0],
                          (unsigned long long)(size_t)(Bm + (size_t)t * 2048));
        }
    };
    if (wave == 0) issue_group(0, 0);
    for (int g = 0; g < NGROUPS; ++g) {
        if (wave == 0) {
            if (g + 1 < NGROUPS) {
                issue_group(g + 1, (g + 1) & 1);
                __builtin_amdgcn_s_wait_tensorcnt(4);   // oldest 4 (group g) done
            } else {
                __builtin_amdgcn_s_wait_tensorcnt(0);
            }
        }
        __syncthreads();                                // group g visible to all
        #pragma unroll
        for (int i = 0; i < 4; ++i)
            consume_tile(&sbuf[g & 1][i][0]);
        __syncthreads();                                // done reading buffer
    }
#else
    // Fallback: cooperative synchronous staging (same padded layout).
    const int row = threadIdx.x >> 4, cir = threadIdx.x & 15;   // 256 chunks/tile
    for (int g = 0; g < NGROUPS; ++g) {
        #pragma unroll
        for (int i = 0; i < 4; ++i) {
            const int t = g * 4 + i;
            const uint4* src = (const uint4*)(Bm + (size_t)t * 2048);
            *(uint4*)&sbuf[0][i][row * 272 + cir * 16] = src[threadIdx.x];
        }
        __syncthreads();
        #pragma unroll
        for (int i = 0; i < 4; ++i)
            consume_tile(&sbuf[0][i][0]);
        __syncthreads();
    }
#endif

    // Recompute the single diagonal tile (x0 == n0) straight from global.
    float draw[8];
    {
        v8f acc = {0.f,0.f,0.f,0.f,0.f,0.f,0.f,0.f};
        const __bf16* Bg = Bm + (size_t)(n0 + l15) * Hdim + 16 * half;
        #pragma unroll
        for (int kc = 0; kc < 4; ++kc) {
            v16bf bv = *(const v16bf*)(Bg + kc * 32);
            acc = __builtin_amdgcn_wmma_f32_16x16x32_bf16(
                      false, areg[kc], false, bv, (short)0, acc, false, false);
        }
        #pragma unroll
        for (int j = 0; j < 8; ++j) draw[j] = acc[j];
    }

    // Merge the 16 lane-local (m,s) states per row; diagonal lane writes out.
    #pragma unroll
    for (int j = 0; j < 8; ++j) {
        float m = m_l[j], s = s_l[j];
        #pragma unroll
        for (int mask = 1; mask < 16; mask <<= 1) {
            const float mo = __shfl_xor(m, mask);
            const float so = __shfl_xor(s, mask);
            const float mn = fmaxf(m, mo);
            s = s * __expf(m - mn) + so * __expf(mo - mn);
            m = mn;
        }
        if (dlane[j])
            diag[(size_t)b * Ndim + n0 + j + 8 * half] =
                __expf(draw[j] - m) / s;
    }
}

// ---------------------------------------------------------------------------
// Kernel 3: out_a = diag * input_b ; out_b = diag * input_a.
// ---------------------------------------------------------------------------
__global__ __launch_bounds__(256) void scale_out_kernel(
    const float* __restrict__ ia, const float* __restrict__ ib,
    const float* __restrict__ diag, float* __restrict__ out)
{
    const size_t CN  = (size_t)Cdim * Ndim;
    const size_t TOT = (size_t)Bsz * CN;
    const size_t i4  = (size_t)blockIdx.x * blockDim.x + threadIdx.x;
    const size_t base = i4 * 4;
    if (base >= TOT) return;
    const int b = (int)(base / CN);
    const int n = (int)(base & (size_t)(Ndim - 1));
    const float4 dv = *(const float4*)(diag + (size_t)b * Ndim + n);
    const float4 av = *(const float4*)(ia + base);
    const float4 bv = *(const float4*)(ib + base);
    float4 oa, ob;
    oa.x = dv.x*bv.x; oa.y = dv.y*bv.y; oa.z = dv.z*bv.z; oa.w = dv.w*bv.w;
    ob.x = dv.x*av.x; ob.y = dv.y*av.y; ob.z = dv.z*av.z; ob.w = dv.w*av.w;
    *(float4*)(out + base)       = oa;
    *(float4*)(out + TOT + base) = ob;
}

// ---------------------------------------------------------------------------
extern "C" void kernel_launch(void* const* d_in, const int* in_sizes, int n_in,
                              void* d_out, int out_size, void* d_ws, size_t ws_size,
                              hipStream_t stream) {
    (void)in_sizes; (void)n_in; (void)out_size; (void)ws_size;
    const float* input_a = (const float*)d_in[0];
    const float* input_b = (const float*)d_in[1];
    const float* Wa      = (const float*)d_in[2];
    const float* ba      = (const float*)d_in[3];
    const float* Wb      = (const float*)d_in[4];
    const float* bb      = (const float*)d_in[5];
    float* out = (float*)d_out;

    char* ws = (char*)d_ws;
    const size_t mappedBytes = (size_t)Bsz * Ndim * Hdim * sizeof(__hip_bfloat16); // 4 MB
    __hip_bfloat16* ma = (__hip_bfloat16*)ws;
    __hip_bfloat16* mb = (__hip_bfloat16*)(ws + mappedBytes);
    float* diag        = (float*)(ws + 2 * mappedBytes);                           // 64 KB

    dim3 mgrid(Ndim / 32, Bsz);
    map_project_kernel<<<mgrid, 256, 0, stream>>>(input_a, Wa, ba, ma);
    map_project_kernel<<<mgrid, 256, 0, stream>>>(input_b, Wb, bb, mb);

    score_diag_kernel<<<Bsz * 32, 256, 0, stream>>>(ma, mb, diag);   // 128 blocks

    const size_t nVec = (size_t)Bsz * Cdim * Ndim / 4;
    scale_out_kernel<<<(unsigned)((nVec + 255) / 256), 256, 0, stream>>>(
        input_a, input_b, diag, out);
}
